// MSMAttention_26920855011864
// MI455X (gfx1250) — matrix operand order repaired and checked by reading, not compiled, not run on hardware
//
#include <hip/hip_runtime.h>

// ---------------------------------------------------------------------------
// Causal MHA block for MI455X (gfx1250, wave32, WMMA f16 16x16x32, f32 accum)
// B=2, T=2048, C=1024, H=16, D=64
// Pipeline: cvt/transpose -> QKV GEMM (WMMA) -> flash attention (WMMA) ->
//           output GEMM (WMMA).  All intermediates f16 in workspace (~48 MB,
//           resident in the 192 MB L2).
// ---------------------------------------------------------------------------

typedef __attribute__((ext_vector_type(16))) _Float16 v16h;
typedef __attribute__((ext_vector_type(8)))  _Float16 v8h;
typedef __attribute__((ext_vector_type(8)))  float    v8f;

#define BB 2
#define TT 2048
#define CCH 1024
#define HH 16
#define DD 64
#define N3 3072
#define MM (BB * TT)   // 4096 rows

static __device__ __forceinline__ v8f zero8() {
  v8f z = {0.f, 0.f, 0.f, 0.f, 0.f, 0.f, 0.f, 0.f};
  return z;
}

// A-fragment for 16-bit WMMA A (16x32): lane<16 owns K {0..15},
// lane>=16 owns K {16..31}. Caller passes the two 8-half chunks.
static __device__ __forceinline__ v16h load_a_frag(const _Float16* p0,
                                                   const _Float16* p1) {
  v8h lo = *(const v8h*)p0;
  v8h hv = *(const v8h*)p1;
  v16h r;
#pragma unroll
  for (int i = 0; i < 8; ++i) { r[i] = lo[i]; r[i + 8] = hv[i]; }
  return r;
}

static __device__ __forceinline__ v8f wmma16(v16h a, v16h b, v8f c) {
  // (neg_a, A, neg_b, B, c_mod, C, reuse_a, reuse_b)
  return __builtin_amdgcn_wmma_f32_16x16x32_f16(false, a, false, b,
                                                (short)0, c, false, false);
}

// ---------------------------------------------------------------------------
// Conversion kernels
// ---------------------------------------------------------------------------
__global__ void cvt_f32_to_f16(const float* __restrict__ in,
                               _Float16* __restrict__ out, int n) {
  int i = blockIdx.x * blockDim.x + threadIdx.x;
  if (i < n) out[i] = (_Float16)in[i];
}

// out[n*K + k] = (f16) in[k*N + n]   (transpose KxN -> NxK)
__global__ void transpose_f32_to_f16(const float* __restrict__ in,
                                     _Float16* __restrict__ out, int K, int N) {
  int i = blockIdx.x * blockDim.x + threadIdx.x;
  if (i < K * N) {
    int k = i % K;
    int n = i / K;
    out[i] = (_Float16)in[(long)k * N + n];
  }
}

// ---------------------------------------------------------------------------
// QKV projection GEMM: qkv = xh @ Wqkv + bqkv, scattered into Q/K/Vt (f16)
//   Q,K : [B,H,T,D] row-major      Vt : [B,H,D,T]
// One wave computes a 32(M) x 64(N) tile (B frags reused across 2 M-subtiles);
// K loop step 32, with prefetch of the next chunk.
// ---------------------------------------------------------------------------
__global__ void __launch_bounds__(256) gemm_qkv_wmma(
    const _Float16* __restrict__ xh, const _Float16* __restrict__ wt,
    const float* __restrict__ bqkv, _Float16* __restrict__ Qf,
    _Float16* __restrict__ Kf, _Float16* __restrict__ Vt) {
  const int lane = threadIdx.x & 31;
  const int wid = blockIdx.x * (blockDim.x >> 5) + (threadIdx.x >> 5);
  const int ntiles = N3 / 64;  // 48
  const int mt = wid / ntiles;
  const int nt = wid % ntiles;
  const int rowBase = mt * 32;
  const int colBase = nt * 64;
  const int l15 = lane & 15, hi = lane >> 4;

  v8f acc[2][4];
#pragma unroll
  for (int m = 0; m < 2; ++m)
#pragma unroll
    for (int t = 0; t < 4; ++t) acc[m][t] = zero8();

  const _Float16* arow0 = xh + (long)(rowBase + l15) * CCH;
  const _Float16* arow1 = arow0 + (long)16 * CCH;
  const _Float16* bcol = wt + (long)(colBase + l15) * CCH + hi * 16;

  for (int kb = 0; kb < CCH; kb += 32) {
    if (kb + 32 < CCH) {
      __builtin_prefetch(arow0 + kb + 32, 0, 1);
      __builtin_prefetch(arow1 + kb + 32, 0, 1);
      __builtin_prefetch(bcol + kb + 32, 0, 1);
    }
    v16h af0 = load_a_frag(arow0 + kb + hi * 16, arow0 + kb + hi * 16 + 8);
    v16h af1 = load_a_frag(arow1 + kb + hi * 16, arow1 + kb + hi * 16 + 8);
#pragma unroll
    for (int t = 0; t < 4; ++t) {
      v16h bf = *(const v16h*)(bcol + (long)t * 16 * CCH + kb);
      acc[0][t] = wmma16(af0, bf, acc[0][t]);
      acc[1][t] = wmma16(af1, bf, acc[1][t]);
    }
  }

  // Epilogue: bias + scatter.  Within this 64-wide tile, s and h are constant;
  // within the 32-row tile, b is constant (2048 % 32 == 0).
  const int s = colBase >> 10;             // 0=Q, 1=K, 2=V
  const int h = (colBase & 1023) >> 6;     // head
  const int b = rowBase >> 11;             // batch
  const long bhQK = (long)(b * HH + h) * TT;
  const long bhV = (long)(b * HH + h) * DD;
#pragma unroll
  for (int t = 0; t < 4; ++t) {
    const int d = t * 16 + l15;
    const float bias = bqkv[colBase + t * 16 + l15];
#pragma unroll
    for (int m = 0; m < 2; ++m) {
#pragma unroll
      for (int r = 0; r < 8; ++r) {
        const int tt = (rowBase & 2047) + m * 16 + r + 8 * hi;
        const _Float16 hv = (_Float16)(acc[m][t][r] + bias);
        if (s == 0) {
          Qf[(bhQK + tt) * DD + d] = hv;
        } else if (s == 1) {
          Kf[(bhQK + tt) * DD + d] = hv;
        } else {
          Vt[(bhV + d) * TT + tt] = hv;
        }
      }
    }
  }
}

// ---------------------------------------------------------------------------
// Flash attention: one wave per (b,h,16-query tile); 32-key chunks.
// S = Q Kt (4 WMMA) -> online softmax (f32) -> P via LDS relayout -> PV (4 WMMA)
// Output written f16 to Yh[B*T][C] with col = h*64 + d.
// ---------------------------------------------------------------------------
__global__ void __launch_bounds__(256) attn_wmma(
    const _Float16* __restrict__ Qf, const _Float16* __restrict__ Kf,
    const _Float16* __restrict__ Vt, _Float16* __restrict__ Yh) {
  __shared__ _Float16 Ps[8 * 512];  // per-wave 16x32 f16 staging (1 KB each)

  const int lane = threadIdx.x & 31;
  const int warp = threadIdx.x >> 5;
  const int wid = blockIdx.x * 8 + warp;
  const int bh = wid >> 7;             // 0..31  (b*16+h)
  const int qBase = (wid & 127) << 4;  // 0..2032
  const int l15 = lane & 15, hi = lane >> 4;
  const long qkBase = (long)bh * TT * DD;
  const long vBase = (long)bh * DD * TT;

  // Q fragments (d 0..31 and 32..63), held for the whole key stream.
  const _Float16* qrow = Qf + qkBase + (long)(qBase + l15) * DD;
  const v16h qa0 = load_a_frag(qrow + hi * 16, qrow + hi * 16 + 8);
  const v16h qa1 = load_a_frag(qrow + 32 + hi * 16, qrow + 40 + hi * 16);

  float mrow[8], lrow[8];
  v8f o0 = zero8(), o1 = zero8(), o2 = zero8(), o3 = zero8();
#pragma unroll
  for (int r = 0; r < 8; ++r) { mrow[r] = -1e30f; lrow[r] = 0.f; }

  _Float16* pw = &Ps[warp * 512];
  const int nCh = (qBase + 47) >> 5;  // ceil((qBase+16)/32) causal chunks

  for (int ch = 0; ch < nCh; ++ch) {
    const int kBase = ch << 5;

    // ---- S = (Q Kt) over d=0..63 : two 16-key column tiles ----
    const _Float16* kp0 = Kf + qkBase + (long)(kBase + l15) * DD + hi * 16;
    const _Float16* kp1 = kp0 + 16 * DD;
    v16h kb00 = *(const v16h*)(kp0);
    v16h kb01 = *(const v16h*)(kp0 + 32);
    v16h kb10 = *(const v16h*)(kp1);
    v16h kb11 = *(const v16h*)(kp1 + 32);
    v8f s0 = zero8();
    s0 = wmma16(qa0, kb00, s0);
    s0 = wmma16(qa1, kb01, s0);
    v8f s1 = zero8();
    s1 = wmma16(qa0, kb10, s1);
    s1 = wmma16(qa1, kb11, s1);

    // ---- scale, causal mask, online softmax ----
    const bool needMask = (kBase + 31 > qBase);
    const int k0 = kBase + l15, k1 = k0 + 16;
    v8f p0, p1;
#pragma unroll
    for (int r = 0; r < 8; ++r) {
      const int q = qBase + r + hi * 8;
      float a0 = s0[r] * 0.125f;  // 1/sqrt(64)
      float a1 = s1[r] * 0.125f;
      if (needMask) {
        if (k0 > q) a0 = -1e30f;
        if (k1 > q) a1 = -1e30f;
      }
      float t = fmaxf(a0, a1);
      t = fmaxf(t, __shfl_xor(t, 1, 16));
      t = fmaxf(t, __shfl_xor(t, 2, 16));
      t = fmaxf(t, __shfl_xor(t, 4, 16));
      t = fmaxf(t, __shfl_xor(t, 8, 16));
      const float mn = fmaxf(mrow[r], t);
      const float sc = __expf(mrow[r] - mn);
      const float e0 = __expf(a0 - mn);
      const float e1 = __expf(a1 - mn);
      float rs = e0 + e1;
      rs += __shfl_xor(rs, 1, 16);
      rs += __shfl_xor(rs, 2, 16);
      rs += __shfl_xor(rs, 4, 16);
      rs += __shfl_xor(rs, 8, 16);
      lrow[r] = lrow[r] * sc + rs;
      mrow[r] = mn;
      p0[r] = e0;
      p1[r] = e1;
      o0[r] *= sc; o1[r] *= sc; o2[r] *= sc; o3[r] *= sc;
    }

    // ---- stage P (C layout -> row-major 16x32 in LDS) ----
#pragma unroll
    for (int r = 0; r < 8; ++r) {
      const int q = r + hi * 8;
      pw[q * 32 + l15] = (_Float16)p0[r];
      pw[q * 32 + 16 + l15] = (_Float16)p1[r];
    }
    asm volatile("s_wait_dscnt 0" ::: "memory");  // same-wave store->load
    const v16h pa =
        load_a_frag(pw + l15 * 32 + hi * 16, pw + l15 * 32 + hi * 16 + 8);

    // ---- O += P V : 4 x 16-d output chunks, K=32 keys ----
    const _Float16* vp = Vt + vBase + (long)l15 * TT + kBase + hi * 16;
    v16h vb0 = *(const v16h*)(vp);
    v16h vb1 = *(const v16h*)(vp + 16 * TT);
    v16h vb2 = *(const v16h*)(vp + 32 * TT);
    v16h vb3 = *(const v16h*)(vp + 48 * TT);
    o0 = wmma16(pa, vb0, o0);
    o1 = wmma16(pa, vb1, o1);
    o2 = wmma16(pa, vb2, o2);
    o3 = wmma16(pa, vb3, o3);
  }

  // ---- finalize and write Y (f16, [B*T][C], col = h*64+d) ----
  const int b = bh >> 4, h = bh & 15;
#pragma unroll
  for (int r = 0; r < 8; ++r) {
    const float inv = 1.0f / lrow[r];
    const long rowG = (long)b * TT + qBase + r + 8 * hi;
    _Float16* yp = Yh + rowG * CCH + h * DD + l15;
    yp[0] = (_Float16)(o0[r] * inv);
    yp[16] = (_Float16)(o1[r] * inv);
    yp[32] = (_Float16)(o2[r] * inv);
    yp[48] = (_Float16)(o3[r] * inv);
  }
}

// ---------------------------------------------------------------------------
// Output projection: out = Yh @ Wo + bo  (f32 result), 32x64 wave tile
// ---------------------------------------------------------------------------
__global__ void __launch_bounds__(256) gemm_out_wmma(
    const _Float16* __restrict__ yh, const _Float16* __restrict__ wot,
    const float* __restrict__ bo, float* __restrict__ out) {
  const int lane = threadIdx.x & 31;
  const int wid = blockIdx.x * (blockDim.x >> 5) + (threadIdx.x >> 5);
  const int ntiles = CCH / 64;  // 16
  const int mt = wid / ntiles;
  const int nt = wid % ntiles;
  const int rowBase = mt * 32;
  const int colBase = nt * 64;
  const int l15 = lane & 15, hi = lane >> 4;

  v8f acc[2][4];
#pragma unroll
  for (int m = 0; m < 2; ++m)
#pragma unroll
    for (int t = 0; t < 4; ++t) acc[m][t] = zero8();

  const _Float16* arow0 = yh + (long)(rowBase + l15) * CCH;
  const _Float16* arow1 = arow0 + (long)16 * CCH;
  const _Float16* bcol = wot + (long)(colBase + l15) * CCH + hi * 16;

  for (int kb = 0; kb < CCH; kb += 32) {
    if (kb + 32 < CCH) {
      __builtin_prefetch(arow0 + kb + 32, 0, 1);
      __builtin_prefetch(arow1 + kb + 32, 0, 1);
      __builtin_prefetch(bcol + kb + 32, 0, 1);
    }
    v16h af0 = load_a_frag(arow0 + kb + hi * 16, arow0 + kb + hi * 16 + 8);
    v16h af1 = load_a_frag(arow1 + kb + hi * 16, arow1 + kb + hi * 16 + 8);
#pragma unroll
    for (int t = 0; t < 4; ++t) {
      v16h bf = *(const v16h*)(bcol + (long)t * 16 * CCH + kb);
      acc[0][t] = wmma16(af0, bf, acc[0][t]);
      acc[1][t] = wmma16(af1, bf, acc[1][t]);
    }
  }

#pragma unroll
  for (int t = 0; t < 4; ++t) {
    const int col = colBase + t * 16 + l15;
    const float bias = bo[col];
#pragma unroll
    for (int m = 0; m < 2; ++m) {
#pragma unroll
      for (int r = 0; r < 8; ++r) {
        const int row = rowBase + m * 16 + r + 8 * hi;
        out[(long)row * CCH + col] = acc[m][t][r] + bias;
      }
    }
  }
}

// ---------------------------------------------------------------------------
// Launch
// ---------------------------------------------------------------------------
extern "C" void kernel_launch(void* const* d_in, const int* in_sizes, int n_in,
                              void* d_out, int out_size, void* d_ws,
                              size_t ws_size, hipStream_t stream) {
  (void)in_sizes; (void)n_in; (void)out_size; (void)ws_size;
  const float* x = (const float*)d_in[0];
  const float* Wqkv = (const float*)d_in[1];
  const float* bqkv = (const float*)d_in[2];
  const float* Wo = (const float*)d_in[3];
  const float* bo = (const float*)d_in[4];
  float* out = (float*)d_out;

  // Workspace layout (f16): ~48 MB total.
  char* w = (char*)d_ws;
  _Float16* xh = (_Float16*)w;    w += (size_t)MM * CCH * 2;
  _Float16* wqkvt = (_Float16*)w; w += (size_t)N3 * CCH * 2;
  _Float16* wot = (_Float16*)w;   w += (size_t)CCH * CCH * 2;
  _Float16* Qf = (_Float16*)w;    w += (size_t)MM * CCH * 2;  // B*H*T*D
  _Float16* Kf = (_Float16*)w;    w += (size_t)MM * CCH * 2;
  _Float16* Vt = (_Float16*)w;    w += (size_t)MM * CCH * 2;  // [B,H,D,T]
  _Float16* Yh = (_Float16*)w;    w += (size_t)MM * CCH * 2;

  cvt_f32_to_f16<<<(MM * CCH + 255) / 256, 256, 0, stream>>>(x, xh, MM * CCH);
  transpose_f32_to_f16<<<(N3 * CCH + 255) / 256, 256, 0, stream>>>(Wqkv, wqkvt,
                                                                   CCH, N3);
  transpose_f32_to_f16<<<(CCH * CCH + 255) / 256, 256, 0, stream>>>(Wo, wot,
                                                                    CCH, CCH);
  // 128x48 = 6144 wave-tiles (32x64 each), 8 waves/block
  gemm_qkv_wmma<<<(MM / 32) * (N3 / 64) / 8, 256, 0, stream>>>(xh, wqkvt, bqkv,
                                                               Qf, Kf, Vt);
  // 2*16*128 = 4096 q-tiles, 8 waves/block
  attn_wmma<<<(BB * HH * (TT / 16)) / 8, 256, 0, stream>>>(Qf, Kf, Vt, Yh);
  // 128x16 = 2048 wave-tiles (32x64 each)
  gemm_out_wmma<<<(MM / 32) * (CCH / 64) / 8, 256, 0, stream>>>(Yh, wot, bo,
                                                                out);
}
